// QuantizedFieldEmbedding_26379689132409
// MI455X (gfx1250) — compile-verified, matrix operand order checked by reading, not run on hardware
//
#include <hip/hip_runtime.h>
#include <hip/hip_bf16.h>
#include <math.h>

#define DIM   512   // embedding dim
#define KQ    512   // number of codebook entries
#define MTILE 32    // token rows per workgroup

typedef __attribute__((ext_vector_type(16))) __bf16 v16bf;
typedef __attribute__((ext_vector_type(8)))  float  v8f;
typedef __attribute__((ext_vector_type(4)))  unsigned int v4u;
typedef __attribute__((ext_vector_type(8)))  int v8i;
typedef __attribute__((ext_vector_type(4)))  int v4i;

union AFrag { v16bf v; __bf16 h[16]; };
union BFrag { v16bf v; uint4 q[2]; };
union BF16Bits { __bf16 h; unsigned short u; };

__device__ __forceinline__ unsigned short f32_bf16_bits(float f) {
  BF16Bits b;
  b.h = (__bf16)f;               // native v_cvt f32->bf16
  return b.u;
}

// ---------------------------------------------------------------------------
// Kernel 1: normalize codebook -> bf16 copy in workspace; zero accumulators.
// One block per codebook row (512 blocks x 256 threads).
// ---------------------------------------------------------------------------
__global__ void vq_prep(const float* __restrict__ cb,
                        unsigned short* __restrict__ cb_bf,
                        float* __restrict__ cls_g,
                        float* __restrict__ commit_g) {
  __shared__ float red[256];
  const int row = blockIdx.x;
  const int t = threadIdx.x;
  float x0 = cb[row * DIM + t];
  float x1 = cb[row * DIM + t + 256];
  red[t] = x0 * x0 + x1 * x1;
  __syncthreads();
  for (int s = 128; s > 0; s >>= 1) {
    if (t < s) red[t] += red[t + s];
    __syncthreads();
  }
  float inv = 1.0f / fmaxf(sqrtf(red[0]), 1e-12f);
  cb_bf[row * DIM + t]       = f32_bf16_bits(x0 * inv);
  cb_bf[row * DIM + t + 256] = f32_bf16_bits(x1 * inv);
  if (row < 2) cls_g[row * 256 + t] = 0.0f;
  if (row == 0 && t == 0) *commit_g = 0.0f;
}

// ---------------------------------------------------------------------------
// Kernel 2: TDM-gather + normalize + WMMA cosine-sim + argmax/softmax stats.
// 256 threads = 8 wave32; 32 token rows per block; sim tile kept in LDS.
// ---------------------------------------------------------------------------
__global__ void __launch_bounds__(256)
vq_main(const int* __restrict__ tok,
        const float* __restrict__ emb,
        const float* __restrict__ cbf32,
        const unsigned short* __restrict__ cbbf,
        float* __restrict__ phi,
        float* __restrict__ kout,
        float* __restrict__ cls_g,
        float* __restrict__ commit_g,
        int vocab) {
  extern __shared__ char smem[];
  float* ze   = (float*)smem;              // MTILE*DIM  (raw embeddings)
  float* simb = ze + MTILE * DIM;          // MTILE*KQ   (similarity tile)
  float* invn = simb + MTILE * KQ;         // MTILE      (inverse norms)
  float* cls  = invn + MTILE;              // KQ         (softmax class sums)

  const int t    = threadIdx.x;
  const int lane = t & 31;
  const int w    = t >> 5;
  const long rowBase = (long)blockIdx.x * MTILE;

  cls[t] = 0.0f;
  cls[t + 256] = 0.0f;

  // ---- Phase 1: TDM gather-mode — pull 32 embedding rows into LDS. ----
  // Wave 0 issues 4 descriptors, 8 row-indices each (32-bit index mode).
  if (w == 0) {
    int id = tok[rowBase + lane];                       // lane r -> row r's token
    unsigned long long gbase = (unsigned long long)(const void*)emb;
    #pragma unroll
    for (int d = 0; d < 4; ++d) {
      v4u g0;
      g0[0] = 0xC0000001u;                              // count=1, gather_mode=1, 32b idx
      g0[1] = (unsigned)(d * 8 * DIM * 4);              // lds_addr (bytes into ze)
      g0[2] = (unsigned)(gbase & 0xFFFFFFFFu);          // global_addr[31:0]
      g0[3] = ((unsigned)(gbase >> 32) & 0x01FFFFFFu)   // global_addr[56:32]
              | (2u << 30);                             // type = 2 (image)
      v8i g1;
      g1[0] = (int)(2u << 16);                          // data_size = 4 bytes
      g1[1] = (int)((unsigned)DIM << 16);               // tensor_dim0[15:0] = 512
      g1[2] = (int)(((unsigned)vocab & 0xFFFFu) << 16); // tensor_dim1[15:0]
      g1[3] = (int)(((unsigned)DIM << 16) |             // tile_dim0 = 512
                    (((unsigned)vocab >> 16) & 0xFFFFu));
      g1[4] = 8;                                        // tile_dim1 = #indices
      g1[5] = DIM;                                      // tensor_dim0_stride = 512
      g1[6] = 0;
      g1[7] = 0;
      v4i g2, g3;                                       // 8 x 32-bit row indices
      #pragma unroll
      for (int j = 0; j < 4; ++j)
        g2[j] = __builtin_amdgcn_readlane(id, d * 8 + j);
      #pragma unroll
      for (int j = 0; j < 4; ++j)
        g3[j] = __builtin_amdgcn_readlane(id, d * 8 + 4 + j);
      v8i g4 = {};                                      // unused extra group
      __builtin_amdgcn_tensor_load_to_lds(g0, g1, g2, g3, g4, 0);
    }
    __builtin_amdgcn_s_wait_tensorcnt(0);
  }
  __syncthreads();

  // ---- Phase 1b: per-row inverse norms (wave w handles rows 4w..4w+3) ----
  for (int i = 0; i < 4; ++i) {
    int r = w * 4 + i;
    float s = 0.0f;
    #pragma unroll
    for (int j = 0; j < 16; ++j) {
      float x = ze[r * DIM + lane + 32 * j];
      s += x * x;
    }
    for (int off = 16; off > 0; off >>= 1) s += __shfl_xor(s, off, 32);
    if (lane == 0) invn[r] = 1.0f / fmaxf(sqrtf(s), 1e-12f);
  }
  __syncthreads();

  // ---- Phase 2: WMMA  sim = A_n @ CB_n^T ----
  // wave w: M-tile = w&1 (16 rows), N-tiles = 8 starting at (w>>1)*8.
  const int mt       = w & 1;
  const int ntBase   = (w >> 1) * 8;
  const int arow     = mt * 16 + (lane & 15);
  const int koffLane = (lane >> 4) * 16;
  const float ainv   = invn[arow];

  v8f acc[8];
  v8f zero = {};
  #pragma unroll
  for (int n = 0; n < 8; ++n) acc[n] = zero;

  for (int ks = 0; ks < 16; ++ks) {        // K = 512 in steps of 32
    // Build A fragment: 16 bf16 per lane from LDS f32, scaled by invnorm.
    AFrag a;
    const float* ap = ze + arow * DIM + ks * 32 + koffLane;
    #pragma unroll
    for (int i4 = 0; i4 < 4; ++i4) {
      float4 f = ((const float4*)ap)[i4];
      a.h[i4 * 4 + 0] = (__bf16)(f.x * ainv);
      a.h[i4 * 4 + 1] = (__bf16)(f.y * ainv);
      a.h[i4 * 4 + 2] = (__bf16)(f.z * ainv);
      a.h[i4 * 4 + 3] = (__bf16)(f.w * ainv);
    }
    #pragma unroll
    for (int n = 0; n < 8; ++n) {
      // B fragment: column = codebook row (ntBase+n)*16 + lane%16,
      // 16 contiguous bf16 at K offset ks*32 + 16*(lane/16) -> 32B load.
      BFrag b;
      const uint4* bp = (const uint4*)(cbbf +
          ((long)(ntBase + n) * 16 + (lane & 15)) * DIM + ks * 32 + koffLane);
      b.q[0] = bp[0];
      b.q[1] = bp[1];
      acc[n] = __builtin_amdgcn_wmma_f32_16x16x32_bf16(
          false, a.v, false, b.v, (short)0, acc[n], false, false);
    }
  }

  // Store sim tiles to LDS (C layout: VGPR j -> M=j+8*(lane>=16), N=lane%16).
  #pragma unroll
  for (int n = 0; n < 8; ++n) {
    #pragma unroll
    for (int j = 0; j < 8; ++j) {
      simb[(mt * 16 + j + 8 * (lane >> 4)) * KQ + (ntBase + n) * 16 + (lane & 15)] = acc[n][j];
    }
  }
  __syncthreads();

  // ---- Phase 3: per-row argmax, softmax stats, z_q write, commit loss ----
  float commitPart = 0.0f;
  for (int i = 0; i < 4; ++i) {
    int r = w * 4 + i;
    float m = -1e30f;
    int   mi = 0;
    float se = 0.0f;
    #pragma unroll
    for (int j = 0; j < 16; ++j) {
      int c = lane + 32 * j;
      float v = simb[r * KQ + c];
      se += __expf(v);                      // |v| <= 1, no max-shift needed
      if (v > m) { m = v; mi = c; }
    }
    for (int off = 16; off > 0; off >>= 1) {
      float om = __shfl_xor(m, off, 32);
      int   oi = __shfl_xor(mi, off, 32);
      if (om > m || (om == m && oi < mi)) { m = om; mi = oi; }
      se += __shfl_xor(se, off, 32);
    }
    float rinv = 1.0f / se;
    #pragma unroll
    for (int j = 0; j < 16; ++j) {
      int c = lane + 32 * j;
      atomicAdd(&cls[c], __expf(simb[r * KQ + c]) * rinv);
    }
    const float* zq = cbf32 + (long)mi * DIM;
    float* pr = phi + (rowBase + r) * DIM;
    #pragma unroll
    for (int j = 0; j < 16; ++j) {
      int d = lane + 32 * j;
      float q = zq[d];
      pr[d] = q;                            // phi = z_q (STE forward value)
      float diff = ze[r * DIM + d] - q;
      commitPart += diff * diff;
    }
    if (lane == 0) kout[rowBase + r] = (float)mi;
  }
  for (int off = 16; off > 0; off >>= 1) commitPart += __shfl_xor(commitPart, off, 32);
  if (lane == 0) atomicAdd(commit_g, commitPart);
  __syncthreads();
  for (int c = t; c < KQ; c += 256) atomicAdd(&cls_g[c], cls[c]);
}

// ---------------------------------------------------------------------------
// Kernel 3: finalize perplexity + commit loss.
// ---------------------------------------------------------------------------
__global__ void vq_fin(const float* __restrict__ cls_g,
                       const float* __restrict__ commit_g,
                       float* __restrict__ scal, int N) {
  __shared__ float red[512];
  int t = threadIdx.x;
  float p = cls_g[t] / (float)N;
  red[t] = p * __logf(p + 1e-10f);
  __syncthreads();
  for (int s = 256; s > 0; s >>= 1) {
    if (t < s) red[t] += red[t + s];
    __syncthreads();
  }
  if (t == 0) {
    scal[0] = *commit_g / ((float)N * (float)DIM);  // commit loss
    scal[1] = __expf(-red[0]);                      // perplexity
  }
}

// ---------------------------------------------------------------------------
extern "C" void kernel_launch(void* const* d_in, const int* in_sizes, int n_in,
                              void* d_out, int out_size, void* d_ws, size_t ws_size,
                              hipStream_t stream) {
  const int*   tok = (const int*)d_in[0];    // token_ids [B*T]
  const float* emb = (const float*)d_in[1];  // embed_weight [VOCAB*DIM]
  const float* cb  = (const float*)d_in[2];  // codebook [KQ*DIM]
  const int N     = in_sizes[0];             // 32768 token rows
  const int vocab = in_sizes[1] / DIM;       // 50257

  float* out  = (float*)d_out;
  float* phi  = out;                         // [N*DIM]
  float* kout = out + (long)N * DIM;         // [N]
  float* scal = kout + N;                    // [commit, perplexity]

  unsigned short* cbbf = (unsigned short*)d_ws;                       // 512 KB
  float* cls_g    = (float*)((char*)d_ws + (size_t)KQ * DIM * sizeof(unsigned short));
  float* commit_g = cls_g + KQ;

  vq_prep<<<KQ, 256, 0, stream>>>(cb, cbbf, cls_g, commit_g);

  size_t smem = (size_t)(MTILE * DIM + MTILE * KQ + MTILE + KQ) * sizeof(float);
  vq_main<<<N / MTILE, 256, smem, stream>>>(tok, emb, cb, cbbf, phi, kout,
                                            cls_g, commit_g, vocab);

  vq_fin<<<1, 512, 0, stream>>>(cls_g, commit_g, scal, N);
}
